// _JetMoeMLP_85787676770835
// MI455X (gfx1250) — matrix operand order repaired and checked by reading, not compiled, not run on hardware
//
#include <hip/hip_runtime.h>

// ---- CDNA5 WMMA fragment types (wave32) ----
typedef __attribute__((ext_vector_type(16))) __bf16 v16bf;
typedef __attribute__((ext_vector_type(8)))  __bf16 v8bf;
typedef __attribute__((ext_vector_type(8)))  float  v8f;

#define HIDDEN 2048
#define FFN    5632
#define NEXP   8
#define T_TOK  4096

// ---------------- helpers ----------------
__device__ __forceinline__ float fast_rcp(float x) {
#if __has_builtin(__builtin_amdgcn_rcpf)
    return __builtin_amdgcn_rcpf(x);
#else
    return 1.f / x;
#endif
}

// 32-byte global -> LDS copy. Async (ASYNCcnt) on CDNA5; plain copy fallback.
__device__ __forceinline__ void copy32B_to_lds(__bf16* l, const __bf16* g) {
#ifndef NO_ASYNC_LDS
    uint32_t lo = (uint32_t)(uintptr_t)l;   // low 32 bits of generic LDS addr = LDS offset
    asm volatile("global_load_async_to_lds_b128 %0, %1, off"           :: "v"(lo), "v"(g) : "memory");
    asm volatile("global_load_async_to_lds_b128 %0, %1, off offset:16" :: "v"(lo), "v"(g) : "memory");
#else
    *(uint4*)l       = *(const uint4*)g;
    *((uint4*)l + 1) = *((const uint4*)g + 1);
#endif
}
__device__ __forceinline__ void async_wait_all() {
#ifndef NO_ASYNC_LDS
    asm volatile("s_wait_asynccnt 0x0" ::: "memory");
#endif
}

// fragment builders, contiguous k order (same for A and B)
//   A 16x32: half h=(lane>>4): K run [16h,16h+16), row=lane&15
//   B 32x16: half h: K run [16h,16h+16), col=lane&15
__device__ __forceinline__ v16bf load_a_frag(const __bf16* rowptr, int lane) {
    return *(const v16bf*)(rowptr + (lane >> 4) * 16);
}
__device__ __forceinline__ v16bf load_b_frag(const __bf16* rowptr, int lane) {
    return *(const v16bf*)(rowptr + (lane >> 4) * 16);
}

#define WMMA_BF16(a, b, c) \
    __builtin_amdgcn_wmma_f32_16x16x32_bf16(false, (a), false, (b), (short)0, (c), false, false)

// ======================================================================
// init: out[t,d] = bias[d]; zero expert counters
// ======================================================================
__global__ void moe_init(float* __restrict__ out, const float* __restrict__ bias,
                         int* __restrict__ counts) {
    int i = blockIdx.x * blockDim.x + threadIdx.x;
    out[i] = bias[i & (HIDDEN - 1)];
    if (i < NEXP) counts[i] = 0;
}

// ======================================================================
// router: logits -> top-2 -> softmax -> gathered per-expert token lists
// ======================================================================
__global__ void moe_router(const float* __restrict__ x, const float* __restrict__ gw,
                           int* __restrict__ counts, int* __restrict__ tok,
                           float* __restrict__ wt) {
    int t = blockIdx.x * blockDim.x + threadIdx.x;
    if (t >= T_TOK) return;
    float acc[NEXP];
#pragma unroll
    for (int e = 0; e < NEXP; ++e) acc[e] = 0.f;
    const float* xr = x + (size_t)t * HIDDEN;
    for (int d = 0; d < HIDDEN; ++d) {
        float xv = xr[d];
#pragma unroll
        for (int e = 0; e < NEXP; ++e) acc[e] += xv * gw[e * HIDDEN + d];  // uniform -> s_load
    }
    float b1 = -__builtin_inff(), b2 = -__builtin_inff();
    int   i1 = 0, i2 = 0;
#pragma unroll
    for (int e = 0; e < NEXP; ++e) {  // strict > : lowest index wins ties (lax.top_k order)
        float v = acc[e];
        if (v > b1)      { b2 = b1; i2 = i1; b1 = v; i1 = e; }
        else if (v > b2) { b2 = v;  i2 = e; }
    }
    float e2 = __expf(b2 - b1);
    float w1 = fast_rcp(1.f + e2);
    float w2 = e2 * w1;
    int p1 = atomicAdd(&counts[i1], 1);
    tok[i1 * T_TOK + p1] = t;  wt[i1 * T_TOK + p1] = w1;
    int p2 = atomicAdd(&counts[i2], 1);
    tok[i2 * T_TOK + p2] = t;  wt[i2 * T_TOK + p2] = w2;
}

// ======================================================================
// in-projection: H[p, f] = silu(x.Wg^T) * (x.Wu^T) * route_w   (bf16 out)
// M = count, N = FFN, K = HIDDEN. Block tile 128x128, KC = 32, double-buffered.
// ======================================================================
__launch_bounds__(256)
__global__ void moe_gemm_in(const float* __restrict__ x,
                            const float* __restrict__ wg,
                            const float* __restrict__ wu,
                            const int*   __restrict__ tok,
                            const float* __restrict__ wt,
                            const int*   __restrict__ cnt,
                            __bf16* __restrict__ H) {
    const int count = *cnt;
    const int m0 = blockIdx.y * 128;
    if (m0 >= count) return;                      // block-uniform
    const int n0 = blockIdx.x * 128;

    __shared__ __bf16 As[2][128][32];             // 16 KB
    __shared__ __bf16 Bg[2][128][32];             // 16 KB
    __shared__ __bf16 Bu[2][128][32];             // 16 KB
    __shared__ int   toks[128];
    __shared__ float wts[128];

    const int tid = threadIdx.x;
    if (tid < 128) {
        int m = m0 + tid;
        toks[tid] = (m < count) ? tok[m] : tok[0];
        wts[tid]  = (m < count) ? wt[m] : 0.f;
    }
    __syncthreads();

    const int wid  = tid >> 5;
    const int lane = tid & 31;
    const int wm = (wid >> 2) * 64;               // 0, 64  (wave covers 64 rows)
    const int wn = (wid & 3) * 32;                // 0..96  (wave covers 32 cols)
    const int lr = lane & 15;

    // stage mapping: 128x32 tile, 16 f32 per thread
    const int s_r = tid >> 1, s_k = (tid & 1) * 16;

    v8f accG[4][2] = {{{},{}},{{},{}},{{},{}},{{},{}}};
    v8f accU[4][2] = {{{},{}},{{},{}},{{},{}},{{},{}}};

#define STAGE_IN(buf, kk)                                                      \
    {                                                                          \
        const float* sx = x + (size_t)toks[s_r] * HIDDEN + (kk) + s_k;         \
        const float* sg = wg + (size_t)(n0 + s_r) * HIDDEN + (kk) + s_k;       \
        const float* su = wu + (size_t)(n0 + s_r) * HIDDEN + (kk) + s_k;       \
        _Pragma("unroll")                                                      \
        for (int j = 0; j < 16; ++j) {                                         \
            As[buf][s_r][s_k + j] = (__bf16)sx[j];                             \
            Bg[buf][s_r][s_k + j] = (__bf16)sg[j];                             \
            Bu[buf][s_r][s_k + j] = (__bf16)su[j];                             \
        }                                                                      \
    }

    STAGE_IN(0, 0);
    __syncthreads();

    const int NK = HIDDEN / 32;                   // 64 chunks
    for (int i = 0; i < NK; ++i) {
        const int cur = i & 1;
        if (i + 1 < NK) STAGE_IN(cur ^ 1, (i + 1) * 32);

        v16bf g0 = load_b_frag(&Bg[cur][wn      + lr][0], lane);
        v16bf g1 = load_b_frag(&Bg[cur][wn + 16 + lr][0], lane);
        v16bf u0 = load_b_frag(&Bu[cur][wn      + lr][0], lane);
        v16bf u1 = load_b_frag(&Bu[cur][wn + 16 + lr][0], lane);
#pragma unroll
        for (int mt = 0; mt < 4; ++mt) {
            v16bf a = load_a_frag(&As[cur][wm + mt * 16 + lr][0], lane);
            accG[mt][0] = WMMA_BF16(a, g0, accG[mt][0]);
            accG[mt][1] = WMMA_BF16(a, g1, accG[mt][1]);
            accU[mt][0] = WMMA_BF16(a, u0, accU[mt][0]);
            accU[mt][1] = WMMA_BF16(a, u1, accU[mt][1]);
        }
        __syncthreads();
    }
#undef STAGE_IN

    // epilogue: h = silu(g)*u*route_w -> H (bf16). C: row=r+8*(lane>>4), col=lane&15.
#pragma unroll
    for (int mt = 0; mt < 4; ++mt)
#pragma unroll
    for (int nt = 0; nt < 2; ++nt) {
        const int col = n0 + wn + nt * 16 + lr;
#pragma unroll
        for (int r = 0; r < 8; ++r) {
            int lm = wm + mt * 16 + (lane >> 4) * 8 + r;
            if (m0 + lm < count) {
                float g = accG[mt][nt][r];
                float u = accU[mt][nt][r];
                float h = g * fast_rcp(1.f + __expf(-g)) * u * wts[lm];
                H[(size_t)(m0 + lm) * FFN + col] = (__bf16)h;
            }
        }
    }
}

// ======================================================================
// out-projection: out[tok[p], d] += H[p, :] . Wo[d, :]
// M = count, N = HIDDEN, K = FFN. Block tile 128x128, KC = 32, double-buffered.
// H tile staged via async global->LDS DMA (bf16, no conversion needed).
// ======================================================================
__launch_bounds__(256)
__global__ void moe_gemm_out(const __bf16* __restrict__ H,
                             const float*  __restrict__ wo,
                             const int*    __restrict__ tok,
                             const int*    __restrict__ cnt,
                             float* __restrict__ out) {
    const int count = *cnt;
    const int m0 = blockIdx.y * 128;
    if (m0 >= count) return;
    const int n0 = blockIdx.x * 128;

    __shared__ __bf16 Ah[2][128][32];             // 16 KB (async-filled)
    __shared__ __bf16 Bo[2][128][32];             // 16 KB
    __shared__ int   toks[128];

    const int tid = threadIdx.x;
    if (tid < 128) {
        int m = m0 + tid;
        toks[tid] = (m < count) ? tok[m] : 0;
    }
    __syncthreads();

    const int wid  = tid >> 5;
    const int lane = tid & 31;
    const int wm = (wid >> 2) * 64;
    const int wn = (wid & 3) * 32;
    const int lr = lane & 15;

    const int s_r = tid >> 1, s_k = (tid & 1) * 16;   // 128x32 tile, 16 elems/thread

    v8f acc[4][2] = {{{},{}},{{},{}},{{},{}},{{},{}}};

#define STAGE_OUT(buf, kk)                                                     \
    {                                                                          \
        copy32B_to_lds(&Ah[buf][s_r][s_k],                                     \
                       H + (size_t)(m0 + s_r) * FFN + (kk) + s_k);             \
        const float* so = wo + (size_t)(n0 + s_r) * FFN + (kk) + s_k;          \
        _Pragma("unroll")                                                      \
        for (int j = 0; j < 16; ++j) Bo[buf][s_r][s_k + j] = (__bf16)so[j];    \
    }

    STAGE_OUT(0, 0);
    async_wait_all();
    __syncthreads();

    const int NK = FFN / 32;                      // 176 chunks
    for (int i = 0; i < NK; ++i) {
        const int cur = i & 1;
        if (i + 1 < NK) STAGE_OUT(cur ^ 1, (i + 1) * 32);

        v16bf o0 = load_b_frag(&Bo[cur][wn      + lr][0], lane);
        v16bf o1 = load_b_frag(&Bo[cur][wn + 16 + lr][0], lane);
#pragma unroll
        for (int mt = 0; mt < 4; ++mt) {
            v16bf a = load_a_frag(&Ah[cur][wm + mt * 16 + lr][0], lane);
            acc[mt][0] = WMMA_BF16(a, o0, acc[mt][0]);
            acc[mt][1] = WMMA_BF16(a, o1, acc[mt][1]);
        }
        async_wait_all();                         // next-buffer DMA landed
        __syncthreads();
    }
#undef STAGE_OUT

    // scatter-accumulate into out (unique (token,col) per lane; experts serialized)
#pragma unroll
    for (int mt = 0; mt < 4; ++mt)
#pragma unroll
    for (int nt = 0; nt < 2; ++nt) {
        const int col = n0 + wn + nt * 16 + lr;
#pragma unroll
        for (int r = 0; r < 8; ++r) {
            int lm = wm + mt * 16 + (lane >> 4) * 8 + r;
            if (m0 + lm < count) {
                float* p = out + (size_t)toks[lm] * HIDDEN + col;
                *p += acc[mt][nt][r];
            }
        }
    }
}

// ======================================================================
// host launcher
// ======================================================================
extern "C" void kernel_launch(void* const* d_in, const int* in_sizes, int n_in,
                              void* d_out, int out_size, void* d_ws, size_t ws_size,
                              hipStream_t stream) {
    const float* x    = (const float*)d_in[0];  // [2,2048,2048]
    const float* gw   = (const float*)d_in[1];  // [8,2048]
    const float* wig  = (const float*)d_in[2];  // [8,5632,2048]
    const float* wiu  = (const float*)d_in[3];  // [8,5632,2048]
    const float* wo   = (const float*)d_in[4];  // [8,2048,5632]
    const float* bias = (const float*)d_in[5];  // [2048]
    float* out = (float*)d_out;                 // [2,2048,2048]

    char* ws = (char*)d_ws;
    int*    counts = (int*)ws;                                   // 8 ints (256B pad)
    int*    tok    = (int*)(ws + 256);                           // [8][4096]
    float*  wt     = (float*)(ws + 256 + NEXP * T_TOK * 4);      // [8][4096]
    __bf16* H      = (__bf16*)(ws + 256 + 2 * NEXP * T_TOK * 4); // [4096][5632] bf16 (~46MB)

    moe_init<<<dim3((T_TOK * HIDDEN) / 256), 256, 0, stream>>>(out, bias, counts);
    moe_router<<<dim3(T_TOK / 256), 256, 0, stream>>>(x, gw, counts, tok, wt);

    for (int e = 0; e < NEXP; ++e) {
        moe_gemm_in<<<dim3(FFN / 128, T_TOK / 128), 256, 0, stream>>>(
            x, wig + (size_t)e * FFN * HIDDEN, wiu + (size_t)e * FFN * HIDDEN,
            tok + e * T_TOK, wt + e * T_TOK, counts + e, H);
        moe_gemm_out<<<dim3(HIDDEN / 128, T_TOK / 128), 256, 0, stream>>>(
            H, wo + (size_t)e * HIDDEN * FFN, tok + e * T_TOK, counts + e, out);
    }
}
